// TextureExtractor_32504312496377
// MI455X (gfx1250) — matrix operand order repaired and checked by reading, not compiled, not run on hardware
//
#include <hip/hip_runtime.h>

typedef __attribute__((ext_vector_type(16))) _Float16 v16h;
typedef __attribute__((ext_vector_type(8)))  _Float16 h8;
typedef __attribute__((ext_vector_type(4)))  _Float16 h4;
typedef __attribute__((ext_vector_type(8)))  float    v8f;

#define IMG_H 1024
#define IMG_W 1024
#define NIMG  32
#define ROWS  16              // active rows per block
#define HALO  7               // max dr
#define TROWS (ROWS + HALO)   // 23 rows staged in LDS

// offsets in output order [d][angle]: d in {1,3,5,7}, angle in {0,45,90,135}
__constant__ int c_dr[16] = {0,1,1,1, 0,2,3,2, 0,4,5,4, 0,5,7,5};
__constant__ int c_dc[16] = {1,1,0,-1, 3,2,0,-2, 5,4,0,-4, 7,5,0,-5};

// ---------------- kernel 1: per-image min / range via async-to-LDS streaming ----------------
// 1 block / image. Each thread DMAs its own 16B slice of each 4KB chunk into a
// double-buffered LDS stage (global_load_async_to_lds_b128, ASYNCcnt pipelined),
// then reduces from LDS. No cross-thread LDS sharing -> no barriers in the loop.
__global__ __launch_bounds__(256) void k_minmax(const float* __restrict__ x,
                                                float* __restrict__ mm) {
    const int img = blockIdx.x;
    const int tid = threadIdx.x;
    const float* p = x + (size_t)img * IMG_H * IMG_W;

    __shared__ float stage[2][1024];   // two 4KB chunk buffers

    const unsigned l0 = (unsigned)(uintptr_t)&stage[0][tid * 4];
    const unsigned l1 = (unsigned)(uintptr_t)&stage[1][tid * 4];

    float mn =  3.402823466e+38f;
    float mx = -3.402823466e+38f;

    // prologue: chunk 0 -> buffer 0
    {
        unsigned long long g = (unsigned long long)(uintptr_t)(p + tid * 4);
        asm volatile("global_load_async_to_lds_b128 %0, %1, off"
                     :: "v"(l0), "v"(g) : "memory");
    }
    for (int ch = 0; ch < 1024; ++ch) {
        if (ch + 1 < 1024) {
            unsigned ln = (ch & 1) ? l0 : l1;
            unsigned long long g =
                (unsigned long long)(uintptr_t)(p + (size_t)(ch + 1) * 1024 + tid * 4);
            asm volatile("global_load_async_to_lds_b128 %0, %1, off"
                         :: "v"(ln), "v"(g) : "memory");
            asm volatile("s_wait_asynccnt 1" ::: "memory");   // older chunk done
        } else {
            asm volatile("s_wait_asynccnt 0" ::: "memory");
        }
        float4 v = *(const float4*)&stage[ch & 1][tid * 4];
        mn = fminf(mn, fminf(fminf(v.x, v.y), fminf(v.z, v.w)));
        mx = fmaxf(mx, fmaxf(fmaxf(v.x, v.y), fmaxf(v.z, v.w)));
    }

    #pragma unroll
    for (int off = 16; off > 0; off >>= 1) {
        mn = fminf(mn, __shfl_xor(mn, off, 32));
        mx = fmaxf(mx, __shfl_xor(mx, off, 32));
    }
    __shared__ float smn[8], smx[8];
    const int wave = tid >> 5;
    if ((tid & 31) == 0) { smn[wave] = mn; smx[wave] = mx; }
    __syncthreads();
    if (tid == 0) {
        mn = smn[0]; mx = smx[0];
        #pragma unroll
        for (int i = 1; i < 8; ++i) { mn = fminf(mn, smn[i]); mx = fmaxf(mx, smx[i]); }
        float r = mx - mn;
        mm[img * 2 + 0] = mn;
        mm[img * 2 + 1] = (r > 0.0f) ? r : 1.0f;
    }
}

// ---------------- WMMA inner step ----------------
// Each lane feeds offset o = lane&15 (A row o and B column o); the WMMA diagonal
// accumulates sum(diff_o^2). Lane needs diffs only at k in [hi*8, hi*8+24):
//   A[0..7] = d[0..7], A[8..15] = d[16..23], B = A (same d elements, same k order).
// Row-invalid pairs are zeroed for free by aliasing qb onto qa (delta==0).
template <int EDGE>
__device__ __forceinline__ void glcm_step(const _Float16* __restrict__ qaP,
                                          int delta, int ccBase, int hi, v8f& acc) {
    const h8 qa0 = *(const h8*)(qaP);
    const h8 qa1 = *(const h8*)(qaP + 8);
    const h8 qa2 = *(const h8*)(qaP + 16);
    const _Float16* qbP = qaP + delta;     // misaligned by dc -> scalar loads
    h8 qb0, qb1, qb2;
    #pragma unroll
    for (int j = 0; j < 8; ++j) {
        qb0[j] = qbP[j];
        qb1[j] = qbP[8 + j];
        qb2[j] = qbP[16 + j];
    }
    h8 d0 = qa0 - qb0;                     // packed f16, integer-exact
    h8 d1 = qa1 - qb1;
    h8 d2 = qa2 - qb2;
    if (EDGE) {                            // column clipping, first/last block only
        #pragma unroll
        for (int j = 0; j < 8; ++j) {
            d0[j] = ((unsigned)(ccBase + j)      < IMG_W) ? d0[j] : (_Float16)0.0f;
            d1[j] = ((unsigned)(ccBase + 8 + j)  < IMG_W) ? d1[j] : (_Float16)0.0f;
            d2[j] = ((unsigned)(ccBase + 16 + j) < IMG_W) ? d2[j] : (_Float16)0.0f;
        }
    }
    v16h A, B;
    const h8 bl = d0;
    const h8 bh = d2;
    #pragma unroll
    for (int j = 0; j < 8; ++j) {
        A[j]     = d0[j];
        A[j + 8] = d2[j];
        B[j]     = bl[j];
        B[j + 8] = bh[j];
    }
    acc = __builtin_amdgcn_wmma_f32_16x16x32_f16(false, A, false, B,
                                                 (short)0, acc, false, false);
}

// ---------------- kernel 2: WMMA sum of squared diffs at 16 offsets ----------------
__global__ __launch_bounds__(256) void k_contrast(const float* __restrict__ x,
                                                  const float* __restrict__ mm,
                                                  float* __restrict__ partial) {
    const int img = blockIdx.x >> 6;
    const int rb  = blockIdx.x & 63;
    const int r0  = rb * ROWS;

    const float mn    = mm[img * 2 + 0];
    const float inv   = 255.0f / mm[img * 2 + 1];

    __shared__ __align__(16) _Float16 qt[TROWS * IMG_W + 32];  // +pad: halo over-reads
    __shared__ float wsum[8][16];

    // ---- stage + quantize: the 23-row tile is one contiguous span ----
    const int nrows = min(TROWS, IMG_H - r0);
    const float4* src = (const float4*)(x + (size_t)img * IMG_H * IMG_W
                                          + (size_t)r0 * IMG_W);
    const int nq = nrows * (IMG_W / 4);
    for (int i = threadIdx.x; i < nq; i += 256) {
        float4 v = src[i];
        h4 q;
        q[0] = (_Float16)truncf((v.x - mn) * inv);
        q[1] = (_Float16)truncf((v.y - mn) * inv);
        q[2] = (_Float16)truncf((v.z - mn) * inv);
        q[3] = (_Float16)truncf((v.w - mn) * inv);
        *(h4*)&qt[i * 4] = q;
    }
    __syncthreads();

    const int lane = threadIdx.x & 31;
    const int wave = threadIdx.x >> 5;
    const int o    = lane & 15;
    const int hi   = lane >> 4;
    const int s    = hi * 8;
    const int dr   = c_dr[o];
    const int dc   = c_dc[o];

    v8f acc = {};
    #pragma unroll
    for (int rhalf = 0; rhalf < 2; ++rhalf) {
        const int rr = 2 * wave + rhalf;
        const int rowbase = rr * IMG_W;
        const int delta = ((r0 + rr + dr) < IMG_H) ? (dr * IMG_W + dc) : 0;

        glcm_step<1>(&qt[rowbase + s], delta, s + dc, hi, acc);          // cb = 0
        for (int cb = 32; cb <= 960; cb += 32)                           // interior
            glcm_step<0>(&qt[rowbase + cb + s], delta, 0, hi, acc);
        glcm_step<1>(&qt[rowbase + 992 + s], delta, 992 + s + dc, hi, acc);
    }

    // ---- diagonal D[o,o]: o<8 -> VGPR o lane o ; o>=8 -> VGPR o-8 lane o+16 ----
    float sv = acc[0];
    #pragma unroll
    for (int r = 1; r < 8; ++r) sv = ((lane & 7) == r) ? acc[r] : sv;
    if (lane < 8)        wsum[wave][lane]      = sv;
    else if (lane >= 24) wsum[wave][lane - 16] = sv;
    __syncthreads();

    if (threadIdx.x < 16) {
        float tsum = 0.0f;
        #pragma unroll
        for (int w = 0; w < 8; ++w) tsum += wsum[w][threadIdx.x];
        partial[(size_t)blockIdx.x * 16 + threadIdx.x] = tsum;
    }
}

// ---------------- kernel 3: deterministic strip reduction + normalize ----------------
__global__ __launch_bounds__(256) void k_final(const float* __restrict__ partial,
                                               float* __restrict__ out) {
    const int i = blockIdx.x * 256 + threadIdx.x;   // 0..511
    if (i >= NIMG * 16) return;
    const int img = i >> 4;
    const int o   = i & 15;
    float s = 0.0f;
    for (int rb = 0; rb < 64; ++rb)
        s += partial[(size_t)(img * 64 + rb) * 16 + o];
    const int dr  = c_dr[o];
    const int dc  = c_dc[o];
    const int adc = (dc < 0) ? -dc : dc;
    const float cnt = (float)((IMG_H - dr) * (IMG_W - adc));
    out[i] = s / cnt;   // mean squared diff == GLCM contrast
}

extern "C" void kernel_launch(void* const* d_in, const int* in_sizes, int n_in,
                              void* d_out, int out_size, void* d_ws, size_t ws_size,
                              hipStream_t stream) {
    const float* x = (const float*)d_in[0];
    float* ws      = (float*)d_ws;
    float* mm      = ws;          // 64 floats: per-image (min, range)
    float* partial = ws + 64;     // 2048 * 16 floats of strip partials

    k_minmax  <<<NIMG,                  256, 0, stream>>>(x, mm);
    k_contrast<<<NIMG * (IMG_H / ROWS), 256, 0, stream>>>(x, mm, partial);
    k_final   <<<2,                     256, 0, stream>>>(partial, (float*)d_out);
}
